// Attention_9320079032376
// MI455X (gfx1250) — matrix operand order repaired and checked by reading, not compiled, not run on hardware
//
#include <hip/hip_runtime.h>

// ---------------------------------------------------------------------------
// GQA attention block for MI455X (gfx1250, wave32, WMMA, TDM).
// Pipeline: RMSNorm -> bf16 QKV GEMMs (WMMA) -> RoPE -> flash attention
// (TDM streams K/V tiles into LDS, WMMA QK^T + online softmax + WMMA P*V)
// -> WMMA output GEMM (f32 out). All matmul operands bf16, f32 accumulate.
// ---------------------------------------------------------------------------

#define DIMM   2048
#define HEADS  32
#define KVH    8
#define DHEAD  64
#define BATCH  2
#define SEQ    2048
#define BS     (BATCH * SEQ)      // 4096 tokens
#define DI     (HEADS * DHEAD)    // 2048
#define DKV    (KVH * DHEAD)      // 512
#define RMS_EPS 1.1920929e-07f

typedef __attribute__((ext_vector_type(16))) __bf16 bf16x16;
typedef __attribute__((ext_vector_type(8)))  __bf16 bf16x8;
typedef __attribute__((ext_vector_type(8)))  float  f32x8;
typedef __attribute__((ext_vector_type(4)))  unsigned int u32x4;
typedef __attribute__((ext_vector_type(8)))  int  i32x8;

// ---------------- WMMA fragment helpers (CDNA5 16x16x32 bf16) --------------
// A-fragment: lane L holds row (L&15), K = 16*(L>>4)..+15 contiguous
// -> one 32B load.
__device__ __forceinline__ bf16x16
load_a_frag(const __bf16* base, int ld, int row0, int k0, int lane) {
    const int hi = lane >> 4;
    const int r  = row0 + (lane & 15);
    const __bf16* p = base + (size_t)r * ld + k0 + hi * 16;
    return *(const bf16x16*)p;
}

// B-fragment from Bt[n][k] (contiguous-K rows): lane L holds column (L&15),
// K = 16*(L>>4)..+15 contiguous -> one 32B load.
__device__ __forceinline__ bf16x16
load_b_frag(const __bf16* bt, int ld, int col0, int k0, int lane) {
    const int n = col0 + (lane & 15);
    const __bf16* p = bt + (size_t)n * ld + k0 + ((lane >> 4) << 4);
    return *(const bf16x16*)p;
}

__device__ __forceinline__ f32x8
wmma_bf16(bf16x16 a, bf16x16 b, f32x8 c) {
    return __builtin_amdgcn_wmma_f32_16x16x32_bf16(
        false, a, false, b, (short)0, c, false, false);
}

// ---------------- TDM: 2-D tile DMA global -> LDS (ISA ch.8 D# layout) -----
// Tensor: rows of tensor_d0 contiguous elements (2B each), row pitch stride0,
// tensor_d1 rows. Tile: tile_d0 x tile_d1 starting at gaddr, packed into LDS
// at byte offset lds_off. Tracked by TENSORcnt.
__device__ __forceinline__ void
tdm_load_2d(unsigned lds_off, const void* gaddr,
            unsigned tensor_d0, unsigned tensor_d1,
            unsigned tile_d0, unsigned tile_d1, unsigned stride0) {
    const unsigned long long ga = (unsigned long long)(size_t)gaddr;
    u32x4 g0;
    g0[0] = 1u;                                            // count=1, no gather
    g0[1] = lds_off;                                       // lds_addr
    g0[2] = (unsigned)(ga & 0xffffffffu);                  // global_addr lo
    g0[3] = (unsigned)((ga >> 32) & 0x01ffffffu) | (2u << 30); // hi + type=2
    i32x8 g1;
    g1[0] = (int)(1u << 16);                               // data_size = 2B
    g1[1] = (int)(tensor_d0 << 16);                        // tensor_dim0 lo
    g1[2] = (int)((tensor_d0 >> 16) | (tensor_d1 << 16));  // dim0 hi | dim1 lo
    g1[3] = (int)((tensor_d1 >> 16) | (tile_d0 << 16));    // dim1 hi | tile0
    g1[4] = (int)(tile_d1 & 0xffffu);                      // tile1, tile2=0
    g1[5] = (int)stride0;                                  // dim0_stride lo
    g1[6] = 0;                                             // stride hi, dim1_stride
    g1[7] = 0;
    asm volatile("tensor_load_to_lds %0, %1" :: "s"(g0), "s"(g1) : "memory");
}

__device__ __forceinline__ unsigned lds_off_of(const void* p) {
    return (unsigned)(size_t)p;   // LDS aperture: addr[31:0] = LDS byte offset
}

// ---------------- 1) weight convert + transpose: f32[K][N] -> bf16[N][K] ---
__global__ __launch_bounds__(256) void k_wt(const float* __restrict__ w,
                                            __bf16* __restrict__ wt,
                                            int K, int N) {
    size_t idx = (size_t)blockIdx.x * blockDim.x + threadIdx.x;
    if (idx >= (size_t)K * N) return;
    int n = (int)(idx % N);
    int kk = (int)(idx / N);
    wt[(size_t)n * K + kk] = (__bf16)w[idx];
}

// ---------------- 2) RMSNorm: f32 row -> bf16 row --------------------------
__global__ __launch_bounds__(256) void k_rmsnorm(const float* __restrict__ x,
                                                 const float* __restrict__ wn,
                                                 __bf16* __restrict__ xn) {
    __shared__ float red[8];
    const int row = blockIdx.x;
    const int tid = threadIdx.x;
    const float* xr = x + (size_t)row * DIMM;
    const float4* x4 = (const float4*)(xr + tid * 8);
    float4 a = x4[0], b = x4[1];
    float ss = a.x*a.x + a.y*a.y + a.z*a.z + a.w*a.w
             + b.x*b.x + b.y*b.y + b.z*b.z + b.w*b.w;
#pragma unroll
    for (int m = 16; m >= 1; m >>= 1) ss += __shfl_xor(ss, m, 32);
    if ((tid & 31) == 0) red[tid >> 5] = ss;
    __syncthreads();
    if (tid < 8) {
        float t = red[tid];
#pragma unroll
        for (int m = 4; m >= 1; m >>= 1) t += __shfl_xor(t, m, 32);
        if (tid == 0) red[0] = t;
    }
    __syncthreads();
    const float rs = rsqrtf(red[0] * (1.0f / DIMM) + RMS_EPS);
    const float* wr = wn + tid * 8;
    __bf16* o = xn + (size_t)row * DIMM + tid * 8;
#pragma unroll
    for (int i = 0; i < 4; ++i) o[i]     = (__bf16)(((const float*)&a)[i] * rs * wr[i]);
#pragma unroll
    for (int i = 0; i < 4; ++i) o[4 + i] = (__bf16)(((const float*)&b)[i] * rs * wr[4 + i]);
}

// ---------------- 3) generic WMMA GEMM -------------------------------------
// C[M,N] = A[M,K](bf16 rm) x Bt[N,K](bf16 contiguous-K). 8 waves/block; each
// wave computes 32(M) x 64(N): two A-fragments share every B-fragment
// (24 B loaded per WMMA). Prefetch runs 4 K-steps ahead.
// mode 0: bf16 [b,h,s,d]   mode 1: bf16 [b,h,d,s]   mode 2: f32 [M,N]
__global__ __launch_bounds__(256) void k_gemm(const __bf16* __restrict__ A,
                                              const __bf16* __restrict__ Bt,
                                              void* __restrict__ Out,
                                              int M, int N, int K,
                                              int mode, int heads) {
    const int lane = threadIdx.x & 31;
    const int wave = threadIdx.x >> 5;
    const int nBlocks = N >> 6;
    const int bm = blockIdx.x / nBlocks;
    const int bn = blockIdx.x % nBlocks;
    const int mrow = bm * 256 + wave * 32;
    const int ncol = bn * 64;
    if (mrow >= M) return;

    f32x8 acc[2][4] = {};
    for (int kk = 0; kk < K; kk += 32) {
        if (kk + 128 < K) {
            __builtin_prefetch(A + (size_t)(mrow + (lane & 15)) * K + kk + 128, 0, 1);
            __builtin_prefetch(A + (size_t)(mrow + 16 + (lane & 15)) * K + kk + 128, 0, 1);
            __builtin_prefetch(Bt + (size_t)(ncol + (lane & 31)) * K + kk + 128, 0, 1);
        }
        bf16x16 a0 = load_a_frag(A, K, mrow,      kk, lane);
        bf16x16 a1 = load_a_frag(A, K, mrow + 16, kk, lane);
#pragma unroll
        for (int j = 0; j < 4; ++j) {
            bf16x16 b = load_b_frag(Bt, K, ncol + 16 * j, kk, lane);
            acc[0][j] = wmma_bf16(a0, b, acc[0][j]);
            acc[1][j] = wmma_bf16(a1, b, acc[1][j]);
        }
    }

    const int hi = lane >> 4;
    const int nl = lane & 15;
#pragma unroll
    for (int half = 0; half < 2; ++half)
#pragma unroll
        for (int j = 0; j < 4; ++j)
#pragma unroll
            for (int v = 0; v < 8; ++v) {
                const int m = mrow + 16 * half + v + 8 * hi;
                const int n = ncol + 16 * j + nl;
                const float val = acc[half][j][v];
                if (mode == 2) {
                    ((float*)Out)[(size_t)m * N + n] = val;
                } else {
                    const int b = m / SEQ, s = m % SEQ;
                    const int h = n >> 6, d = n & 63;
                    __bf16* ob = (__bf16*)Out;
                    if (mode == 0)
                        ob[(((size_t)b * heads + h) * SEQ + s) * DHEAD + d] = (__bf16)val;
                    else
                        ob[(((size_t)b * heads + h) * DHEAD + d) * SEQ + s] = (__bf16)val;
                }
            }
}

// ---------------- 4) interleaved RoPE on bf16 [b*h][s][64] -----------------
__global__ __launch_bounds__(256) void k_rope(__bf16* __restrict__ t,
                                              int totalPairs) {
    int idx = blockIdx.x * blockDim.x + threadIdx.x;
    if (idx >= totalPairs) return;
    const int i = idx & 31;
    const int rs = idx >> 5;
    const int s = rs & (SEQ - 1);
    __bf16* p = t + (size_t)rs * DHEAD + 2 * i;
    const float t0 = (float)p[0], t1 = (float)p[1];
    const float invf = __powf(10000.0f, -(float)(2 * i) / (float)DHEAD);
    float sn, cs;
    __sincosf((float)s * invf, &sn, &cs);
    p[0] = (__bf16)(t0 * cs - t1 * sn);
    p[1] = (__bf16)(t1 * cs + t0 * sn);
}

// ---------------- 5) causal flash attention (TDM + WMMA) -------------------
// grid = B*HEADS*(SEQ/128); block = 256 = 8 waves; all waves share one head.
// Wave 0 drives the Tensor Data Mover: K tile [32 keys][64 d] and V tile
// [64 d][32 keys] stream into double-buffered LDS (TENSORcnt), all 8 waves
// consume them as WMMA B-fragments via ds loads. Uniform trip count
// (4*(t+1) key blocks) so __syncthreads pipelining is divergence-free.
__global__ __launch_bounds__(256) void k_attn(const __bf16* __restrict__ qb,
                                              const __bf16* __restrict__ kb,
                                              const __bf16* __restrict__ vtb,
                                              __bf16* __restrict__ ob) {
    __shared__ __align__(16) __bf16 ktile[2][32][64];   // 8 KB
    __shared__ __align__(16) __bf16 vtile[2][64][32];   // 8 KB
    __shared__ __align__(16) __bf16 pstage[8][16][32];  // 8 KB

    const int lane = threadIdx.x & 31;
    const int wave = threadIdx.x >> 5;
    const int tilesPerHead = SEQ / 128;
    const int t  = blockIdx.x % tilesPerHead;
    const int bh = blockIdx.x / tilesPerHead;
    const int b  = bh / HEADS;
    const int h  = bh % HEADS;
    const int kvh = h % KVH;

    const __bf16* qh = qb  + ((size_t)(b * HEADS + h)) * SEQ * DHEAD;
    const __bf16* kh = kb  + ((size_t)(b * KVH + kvh)) * SEQ * DHEAD;   // [s][d]
    const __bf16* vh = vtb + ((size_t)(b * KVH + kvh)) * DHEAD * SEQ;   // [d][s]

    const int qrow = t * 128 + wave * 16;
    const int hi = lane >> 4;
    const int nl = lane & 15;

    const bf16x16 aq0 = load_a_frag(qh, DHEAD, qrow, 0,  lane);
    const bf16x16 aq1 = load_a_frag(qh, DHEAD, qrow, 32, lane);

    f32x8 of0 = {}, of1 = {}, of2 = {}, of3 = {};
    float mrow[8], lrow[8];
#pragma unroll
    for (int v = 0; v < 8; ++v) { mrow[v] = -3.0e38f; lrow[v] = 0.0f; }

    const float scale = 0.125f;          // 64^-0.5
    const int kend = qrow + 16;          // this wave's causal horizon
    const int nblk = 4 * (t + 1);        // uniform across the block

    if (wave == 0) {
        tdm_load_2d(lds_off_of(&ktile[0][0][0]), kh, DHEAD, SEQ, DHEAD, 32, DHEAD);
        tdm_load_2d(lds_off_of(&vtile[0][0][0]), vh, SEQ, DHEAD, 32, DHEAD, SEQ);
    }

    for (int i = 0; i < nblk; ++i) {
        const int ks = 32 * i;
        const int buf = i & 1;
        if (wave == 0) __builtin_amdgcn_s_wait_tensorcnt(0);
        __syncthreads();                 // tile i visible; tile i-1 fully read
        if (wave == 0 && i + 1 < nblk) { // prefetch tile i+1 into other buffer
            const int ks2 = ks + 32;
            tdm_load_2d(lds_off_of(&ktile[buf ^ 1][0][0]),
                        kh + (size_t)ks2 * DHEAD, DHEAD, SEQ, DHEAD, 32, DHEAD);
            tdm_load_2d(lds_off_of(&vtile[buf ^ 1][0][0]),
                        vh + ks2, SEQ, DHEAD, 32, DHEAD, SEQ);
        }
        if (ks >= kend) continue;        // beyond causal horizon: barrier-only

        // ---- scores: Q[16x64] x K^T -> two 16x16 tiles (K/V from LDS)
        f32x8 sc0 = {}, sc1 = {};
        sc0 = wmma_bf16(aq0, *(const bf16x16*)&ktile[buf][nl][16 * hi],       sc0);
        sc0 = wmma_bf16(aq1, *(const bf16x16*)&ktile[buf][nl][32 + 16 * hi],  sc0);
        sc1 = wmma_bf16(aq0, *(const bf16x16*)&ktile[buf][16 + nl][16 * hi],      sc1);
        sc1 = wmma_bf16(aq1, *(const bf16x16*)&ktile[buf][16 + nl][32 + 16 * hi], sc1);

        // ---- scale + causal mask + block row-max
        float bmax[8];
#pragma unroll
        for (int v = 0; v < 8; ++v) {
            const int qr = qrow + v + 8 * hi;
            float s0 = sc0[v] * scale;
            float s1 = sc1[v] * scale;
            if (ks + nl      > qr) s0 = -3.0e38f;
            if (ks + 16 + nl > qr) s1 = -3.0e38f;
            sc0[v] = s0; sc1[v] = s1;
            bmax[v] = fmaxf(s0, s1);
        }
#pragma unroll
        for (int m = 8; m >= 1; m >>= 1)
#pragma unroll
            for (int v = 0; v < 8; ++v)
                bmax[v] = fmaxf(bmax[v], __shfl_xor(bmax[v], m, 32));

        // ---- online softmax update
        float alpha[8], rsum[8];
#pragma unroll
        for (int v = 0; v < 8; ++v) {
            const float mn = fmaxf(mrow[v], bmax[v]);
            alpha[v] = __expf(mrow[v] - mn);
            mrow[v] = mn;
            const float p0 = (sc0[v] > -1.0e37f) ? __expf(sc0[v] - mn) : 0.0f;
            const float p1 = (sc1[v] > -1.0e37f) ? __expf(sc1[v] - mn) : 0.0f;
            sc0[v] = p0; sc1[v] = p1;
            rsum[v] = p0 + p1;
        }
#pragma unroll
        for (int m = 8; m >= 1; m >>= 1)
#pragma unroll
            for (int v = 0; v < 8; ++v)
                rsum[v] += __shfl_xor(rsum[v], m, 32);
#pragma unroll
        for (int v = 0; v < 8; ++v) lrow[v] = lrow[v] * alpha[v] + rsum[v];
#pragma unroll
        for (int v = 0; v < 8; ++v) {
            of0[v] *= alpha[v]; of1[v] *= alpha[v];
            of2[v] *= alpha[v]; of3[v] *= alpha[v];
        }

        // ---- stage P (16x32) to LDS, reload as an A-fragment
#pragma unroll
        for (int v = 0; v < 8; ++v) {
            const int m = v + 8 * hi;
            pstage[wave][m][nl]      = (__bf16)sc0[v];
            pstage[wave][m][16 + nl] = (__bf16)sc1[v];
        }
        asm volatile("s_wait_dscnt 0" ::: "memory");
        bf16x16 ap = *(const bf16x16*)&pstage[wave][nl][16 * hi];

        // ---- O += P * V  (V tile [d][keys] in LDS -> contiguous-K B-frags)
        of0 = wmma_bf16(ap, *(const bf16x16*)&vtile[buf][nl][16 * hi],      of0);
        of1 = wmma_bf16(ap, *(const bf16x16*)&vtile[buf][16 + nl][16 * hi], of1);
        of2 = wmma_bf16(ap, *(const bf16x16*)&vtile[buf][32 + nl][16 * hi], of2);
        of3 = wmma_bf16(ap, *(const bf16x16*)&vtile[buf][48 + nl][16 * hi], of3);
    }

    // ---- normalize and scatter O as bf16 rows [token][h*64 + d]
    float inv[8];
#pragma unroll
    for (int v = 0; v < 8; ++v) inv[v] = (lrow[v] > 0.0f) ? 1.0f / lrow[v] : 0.0f;
    f32x8 ofs[4] = {of0, of1, of2, of3};
#pragma unroll
    for (int j = 0; j < 4; ++j)
#pragma unroll
        for (int v = 0; v < 8; ++v) {
            const size_t gm = (size_t)b * SEQ + qrow + v + 8 * hi;
            ob[gm * DI + h * DHEAD + 16 * j + nl] = (__bf16)(ofs[j][v] * inv[v]);
        }
}

// ---------------------------------------------------------------------------
extern "C" void kernel_launch(void* const* d_in, const int* in_sizes, int n_in,
                              void* d_out, int out_size, void* d_ws, size_t ws_size,
                              hipStream_t stream) {
    const float* x   = (const float*)d_in[0];
    const float* wn  = (const float*)d_in[1];
    const float* wq  = (const float*)d_in[2];
    const float* wk  = (const float*)d_in[3];
    const float* wv  = (const float*)d_in[4];
    const float* wo  = (const float*)d_in[5];
    float* out = (float*)d_out;

    char* ws = (char*)d_ws;
    size_t off = 0;
    __bf16* xn   = (__bf16*)(ws + off); off += (size_t)BS * DIMM * 2;
    __bf16* wqt  = (__bf16*)(ws + off); off += (size_t)DI * DIMM * 2;
    __bf16* wkt  = (__bf16*)(ws + off); off += (size_t)DKV * DIMM * 2;
    __bf16* wvt  = (__bf16*)(ws + off); off += (size_t)DKV * DIMM * 2;
    __bf16* wot  = (__bf16*)(ws + off); off += (size_t)DIMM * DI * 2;
    __bf16* qbuf = (__bf16*)(ws + off); off += (size_t)BATCH * HEADS * SEQ * DHEAD * 2;
    __bf16* kbuf = (__bf16*)(ws + off); off += (size_t)BATCH * KVH * SEQ * DHEAD * 2;
    __bf16* vtb  = (__bf16*)(ws + off); off += (size_t)BATCH * KVH * DHEAD * SEQ * 2;
    __bf16* obuf = (__bf16*)(ws + off); off += (size_t)BS * DI * 2;
    (void)off; (void)ws_size; (void)in_sizes; (void)n_in; (void)out_size;

    {
        int n;
        n = DIMM * DI;  k_wt<<<(n + 255) / 256, 256, 0, stream>>>(wq, wqt, DIMM, DI);
        n = DIMM * DKV; k_wt<<<(n + 255) / 256, 256, 0, stream>>>(wk, wkt, DIMM, DKV);
        n = DIMM * DKV; k_wt<<<(n + 255) / 256, 256, 0, stream>>>(wv, wvt, DIMM, DKV);
        n = DI * DIMM;  k_wt<<<(n + 255) / 256, 256, 0, stream>>>(wo, wot, DI, DIMM);
    }

    k_rmsnorm<<<BS, 256, 0, stream>>>(x, wn, xn);

    k_gemm<<<(BS / 256) * (DI / 64),  256, 0, stream>>>(xn, wqt, qbuf, BS, DI,  DIMM, 0, HEADS);
    k_gemm<<<(BS / 256) * (DKV / 64), 256, 0, stream>>>(xn, wkt, kbuf, BS, DKV, DIMM, 0, KVH);
    k_gemm<<<(BS / 256) * (DKV / 64), 256, 0, stream>>>(xn, wvt, vtb,  BS, DKV, DIMM, 1, KVH);

    {
        int nq = BATCH * HEADS * SEQ * (DHEAD / 2);
        int nk = BATCH * KVH   * SEQ * (DHEAD / 2);
        k_rope<<<(nq + 255) / 256, 256, 0, stream>>>(qbuf, nq);
        k_rope<<<(nk + 255) / 256, 256, 0, stream>>>(kbuf, nk);
    }

    k_attn<<<BATCH * HEADS * (SEQ / 128), 256, 0, stream>>>(qbuf, kbuf, vtb, obuf);

    k_gemm<<<(BS / 256) * (DIMM / 64), 256, 0, stream>>>(obuf, wot, out, BS, DIMM, DI, 2, HEADS);
}
